// ResidualCodebookRemapReadout_84928683311888
// MI455X (gfx1250) — matrix operand order repaired and checked by reading, not compiled, not run on hardware
//
#include <hip/hip_runtime.h>
#include <hip/hip_bf16.h>
#include <hip/hip_fp16.h>

typedef __attribute__((ext_vector_type(16))) _Float16 v16h;
typedef __attribute__((ext_vector_type(8)))  _Float16 v8h;
typedef __attribute__((ext_vector_type(4)))  _Float16 v4h;
typedef __attribute__((ext_vector_type(2)))  _Float16 v2h;
typedef __attribute__((ext_vector_type(8)))  float    v8f;
typedef __attribute__((ext_vector_type(4)))  unsigned int u32x4;
typedef __attribute__((ext_vector_type(4)))  int      i32x4;
typedef __attribute__((ext_vector_type(8)))  int      i32x8;

#define WMMA_F16(a,b,c) __builtin_amdgcn_wmma_f32_16x16x32_f16(false,(a),false,(b),(short)0,(c),false,false)

constexpr int Nn = 1024, Dd = 256, Cc = 256, Rr = 64, CLS = 1000, CLSP = 1008;
constexpr int LDH = 272;   // padded f16 LDS row stride (bank spread)
constexpr int RST = 72;    // rs scratch stride

// workspace layout (units: _Float16 elements)
constexpr size_t OFF_CB  = 0;
constexpr size_t OFF_RCB = OFF_CB  + 256*256;
constexpr size_t OFF_WQ  = OFF_RCB + 64*256;
constexpr size_t OFF_WC  = OFF_WQ  + 256*256;
constexpr size_t OFF_WR  = OFF_WC  + (size_t)CLSP*256;
constexpr size_t OFF_M2  = OFF_WR  + 256*256;
constexpr size_t OFF_SUM = OFF_M2  + 64*256;

// LDS layout byte offsets (dynamic shared starts at 0)
constexpr unsigned LDSOFF_CBH  = 0;
constexpr unsigned LDSOFF_RCBH = 256*LDH*2;
constexpr unsigned LDSOFF_M2T  = LDSOFF_RCBH + 64*LDH*2;

constexpr size_t SMEM_BYTES = (size_t)(272*(256+64+64+64+32) + 32*72)*2 + (size_t)(256*5+64+64)*4;

// ---- TDM: async 2D f16 tensor copy global -> LDS with row padding 256 -> 272 halves ----
// pad_interval=6 (128 DWORDs = 512B = one 256-half row), pad_amount=7 (8 DWORDs = 32B = 16 halves)
__device__ __forceinline__ void tdm_load_2d_f16(const _Float16* src, unsigned lds_byte_off, int rows){
    unsigned long long ga = (unsigned long long)src;
    u32x4 g0;
    g0[0] = 1u;                                               // count=1 valid user D#
    g0[1] = lds_byte_off;                                     // lds_addr [63:32]
    g0[2] = (unsigned)ga;                                     // global_addr [95:64]
    g0[3] = (unsigned)((ga >> 32) & 0x1FFFFFFu) | 0x80000000u; // addr[56:32] | type=2
    i32x8 g1;
    g1[0] = (1 << 16) | (1 << 20) | (6 << 22) | (7 << 25);    // data_size=2B, pad_enable, pad_interval, pad_amount
    g1[1] = (int)(256u << 16);                                // tensor_dim0 = 256 (cols)
    g1[2] = (int)((unsigned)rows << 16);                      // tensor_dim1 = rows
    g1[3] = (int)(256u << 16);                                // tile_dim0 = 256
    g1[4] = rows;                                             // tile_dim1 = rows, tile_dim2 = 0
    g1[5] = 256;                                              // tensor_dim0_stride = 256 elements
    g1[6] = 0;
    g1[7] = 0;
    i32x4 gz = {0, 0, 0, 0};
#if defined(__clang_major__) && (__clang_major__ >= 23)
    i32x8 gz8 = {0, 0, 0, 0, 0, 0, 0, 0};
    __builtin_amdgcn_tensor_load_to_lds(g0, g1, gz, gz, gz8, 0);
#else
    __builtin_amdgcn_tensor_load_to_lds(g0, g1, gz, gz, 0);
#endif
}

// ---- WMMA fragment loaders (ISA 7.12.2 layouts); work for LDS or global ptrs ----
__device__ __forceinline__ v16h fragA(const _Float16* base, int stride, int kbase){
    int l = threadIdx.x & 31;
    const _Float16* p = base + (l & 15)*stride + kbase + ((l >> 4) << 3);
    v8h lo = *(const v8h*)p;
    v8h hi = *(const v8h*)(p + 16);
    v16h r;
#pragma unroll
    for (int i = 0; i < 8; i++){ r[i] = lo[i]; r[i+8] = hi[i]; }
    return r;
}
__device__ __forceinline__ v16h fragB(const _Float16* base, int stride, int kbase){
    int l = threadIdx.x & 31;
    const _Float16* p = base + (l & 15)*stride + kbase + ((l >> 4) << 3);
    v8h lo = *(const v8h*)p;
    v8h hi = *(const v8h*)(p + 16);
    v16h r;
#pragma unroll
    for (int i = 0; i < 8; i++){ r[i] = lo[i]; r[i+8] = hi[i]; }
    return r;
}
__device__ __forceinline__ void storeAccH(const v8f acc, _Float16* dst, int stride,
                                          int rowbase, int colbase){
    int l = threadIdx.x & 31;
    int n  = l & 15;
    int mo = (l >> 4) << 3;
#pragma unroll
    for (int v = 0; v < 8; v++)
        dst[(size_t)(rowbase + mo + v)*stride + colbase + n] = (_Float16)acc[v];
}

// ---- block reductions ----
__device__ __forceinline__ float blockMax(float v, float* red){
#pragma unroll
    for (int m = 16; m >= 1; m >>= 1) v = fmaxf(v, __shfl_xor(v, m, 32));
    if ((threadIdx.x & 31) == 0) red[threadIdx.x >> 5] = v;
    __syncthreads();
    float r = red[0];
#pragma unroll
    for (int i = 1; i < 8; i++) r = fmaxf(r, red[i]);
    __syncthreads();
    return r;
}
__device__ __forceinline__ float blockSum(float v, float* red){
#pragma unroll
    for (int m = 16; m >= 1; m >>= 1) v += __shfl_xor(v, m, 32);
    if ((threadIdx.x & 31) == 0) red[threadIdx.x >> 5] = v;
    __syncthreads();
    float r = red[0];
#pragma unroll
    for (int i = 1; i < 8; i++) r += red[i];
    __syncthreads();
    return r;
}

// ---- prep: f32 -> f16 weight conversion ----
__global__ void kPrep(const float* cb, const float* rcb, const float* wq,
                      const float* wc, const float* wr, _Float16* ws){
    size_t idx = (size_t)blockIdx.x*256 + threadIdx.x;
    if (idx < 65536){ ws[OFF_CB + idx] = (_Float16)cb[idx]; return; }
    idx -= 65536;
    if (idx < 16384){ ws[OFF_RCB + idx] = (_Float16)rcb[idx]; return; }
    idx -= 16384;
    if (idx < 65536){ ws[OFF_WQ + idx] = (_Float16)wq[idx]; return; }
    idx -= 65536;
    if (idx < 65536){ ws[OFF_WR + idx] = (_Float16)wr[idx]; return; }
    idx -= 65536;
    if (idx < (size_t)CLSP*256){
        ws[OFF_WC + idx] = (idx < (size_t)CLS*256) ? (_Float16)wc[idx] : (_Float16)0.f;
    }
}

// ---- M2t[r][c] = sum_d codebook[c][d] * rcb[r][d] ----
__global__ void kM2(const float* cb, const float* rcb, _Float16* ws){
    int r = blockIdx.x, c = threadIdx.x;
    const float* a = rcb + (size_t)r*256;
    const float* b = cb  + (size_t)c*256;
    float s = 0.f;
#pragma unroll 8
    for (int d = 0; d < 256; d++) s += a[d]*b[d];
    ws[OFF_M2 + (size_t)r*256 + c] = (_Float16)s;
}

__device__ __forceinline__ void stageKload(const float* Kn, int srow, float4* reg, int t){
    int lr = t >> 3, d0 = (t & 7)*32;
    const float4* p = (const float4*)(Kn + (size_t)(srow + lr)*256 + d0);
#pragma unroll
    for (int i = 0; i < 8; i++) reg[i] = p[i];
}
__device__ __forceinline__ void stageKstore(const float4* reg, _Float16* buf, int t){
    int lr = t >> 3, d0 = (t & 7)*32;
    _Float16* dst = buf + (size_t)lr*LDH + d0;
#pragma unroll
    for (int i = 0; i < 8; i++){
        v4h h; h[0]=(_Float16)reg[i].x; h[1]=(_Float16)reg[i].y;
               h[2]=(_Float16)reg[i].z; h[3]=(_Float16)reg[i].w;
        *(v4h*)(dst + i*4) = h;
    }
}

// ---- main fused kernel: one n per block ----
__global__ __launch_bounds__(256, 1)
void kMain(const float* __restrict__ q, const float* __restrict__ K,
           const float* __restrict__ V, const float* __restrict__ rlogit,
           const _Float16* __restrict__ cbW, const _Float16* __restrict__ rcbW,
           const _Float16* __restrict__ wqW, const _Float16* __restrict__ m2W,
           float* __restrict__ outEff, _Float16* __restrict__ sumOut){
    extern __shared__ char smem[];
    _Float16* cbh  = (_Float16*)smem;           // 256 x LDH
    _Float16* rcbh = cbh  + 256*LDH;            // 64 x LDH
    _Float16* m2t  = rcbh + 64*LDH;             // 64 x LDH
    _Float16* kbuf = m2t  + 64*LDH;             // 2 x 32 x LDH
    _Float16* sch  = kbuf + 2*32*LDH;           // 32 x LDH  (scores -> -scw in place)
    _Float16* rsh  = sch  + 32*LDH;             // 32 x RST
    float* fscr = (float*)(rsh + 32*RST);
    float* qs   = fscr;        float* qp   = qs  + 256;
    float* cwv  = qp  + 256;   float* wscv = cwv + 256;
    float* tmpv = wscv + 256;  float* rcwv = tmpv + 256;
    float* redv = rcwv + 64;

    const int n = blockIdx.x, t = threadIdx.x;
    const int wave = t >> 5;
    const float* Kn = K + (size_t)n*65536;
    const float* Vn = V + (size_t)n*65536;

    // stage constant matrices into LDS via Tensor Data Mover (async DMA, row-padded)
    if (wave == 0)      tdm_load_2d_f16(cbW,  LDSOFF_CBH,  256);
    else if (wave == 1) tdm_load_2d_f16(rcbW, LDSOFF_RCBH, 64);
    else if (wave == 2) tdm_load_2d_f16(m2W,  LDSOFF_M2T,  64);
    qs[t] = q[(size_t)n*256 + t];
    if (wave < 3) __builtin_amdgcn_s_wait_tensorcnt(0);
    __syncthreads();

    // q_proj[t] = sum_e Wq[t][e] * q[e]
    {
        const _Float16* wr = wqW + (size_t)t*256;
        float a = 0.f;
        for (int e = 0; e < 256; e += 8){
            v8h h = *(const v8h*)(wr + e);
#pragma unroll
            for (int j = 0; j < 8; j++) a += (float)h[j]*qs[e + j];
        }
        qp[t] = a;
    }
    __syncthreads();

    // cw = softmax(q_proj @ cb^T / 16)
    {
        const _Float16* cr = cbh + (size_t)t*LDH;
        float a = 0.f;
        for (int d = 0; d < 256; d += 8){
            v8h h = *(const v8h*)(cr + d);
#pragma unroll
            for (int j = 0; j < 8; j++) a += (float)h[j]*qp[d + j];
        }
        tmpv[t] = a;
    }
    __syncthreads();
    {
        float raw = tmpv[t];
        float mx = blockMax(raw, redv);
        float e  = __expf((raw - mx)*0.0625f);
        float s  = blockSum(e, redv);
        cwv[t] = e / s;
    }
    __syncthreads();

    // rcw = softmax((q_proj@rcb^T - cw@M2)/16)
    {
        float raw = -1e30f;
        if (t < 64){
            const _Float16* rr = rcbh + (size_t)t*LDH;
            const _Float16* mr = m2t  + (size_t)t*LDH;
            float a = 0.f, b = 0.f;
            for (int d = 0; d < 256; d += 8){
                v8h h1 = *(const v8h*)(rr + d);
                v8h h2 = *(const v8h*)(mr + d);
#pragma unroll
                for (int j = 0; j < 8; j++){ a += (float)h1[j]*qp[d+j]; b += (float)h2[j]*cwv[d+j]; }
            }
            raw = a - b;
        }
        float mx = blockMax(raw, redv);
        float e  = (t < 64) ? __expf((raw - mx)*0.0625f) : 0.f;
        float s  = blockSum(e, redv);
        if (t < 64) rcwv[t] = e / s;
    }
    const float gate = 1.f/(1.f + __expf(-rlogit[0]));

    // stage K chunk 0 (f32 -> f16 conversion, so VALU path)
    {
        float4 kr[8];
        stageKload(Kn, 0, kr, t);
        stageKstore(kr, kbuf, t);
    }
    __syncthreads();

    const int rt  = wave & 1;       // row tile in 32-row chunk
    const int ctg = wave >> 1;      // 4 c-tiles per wave / r-tile id

    for (int p = 0; p < 8; p++){
        float4 kst[8];
        if (p < 7) stageKload(Kn, (p + 1)*32, kst, t);

        _Float16* kb = kbuf + (size_t)(p & 1)*32*LDH;
        v16h a[8];
#pragma unroll
        for (int k = 0; k < 8; k++) a[k] = fragA(kb + rt*16*LDH, LDH, k*32);

        // GEMM1: scores[32 x 256] = Kchunk @ cb^T
#pragma unroll
        for (int i = 0; i < 4; i++){
            int ct = ctg*4 + i;
            v8f acc = {};
#pragma unroll
            for (int k = 0; k < 8; k++){
                v16h b = fragB(cbh + (size_t)ct*16*LDH, LDH, k*32);
                acc = WMMA_F16(a[k], b, acc);
            }
            storeAccH(acc, sch, LDH, rt*16, ct*16);
        }
        __syncthreads();

        // row softmax -> writes NEGATED scw in place; fused base_weights dot
        {
            int r = t >> 3, j = t & 7, c0 = j*32;
            _Float16* rp = sch + (size_t)r*LDH + c0;
            float x[32];
#pragma unroll
            for (int i = 0; i < 32; i++) x[i] = (float)rp[i];
            float mx = x[0];
#pragma unroll
            for (int i = 1; i < 32; i++) mx = fmaxf(mx, x[i]);
#pragma unroll
            for (int m = 1; m <= 4; m <<= 1) mx = fmaxf(mx, __shfl_xor(mx, m, 32));
            float s = 0.f;
#pragma unroll
            for (int i = 0; i < 32; i++){ x[i] = __expf((x[i] - mx)*0.0625f); s += x[i]; }
#pragma unroll
            for (int m = 1; m <= 4; m <<= 1) s += __shfl_xor(s, m, 32);
            float inv = 1.f / s;
            float bw = 0.f;
#pragma unroll
            for (int i = 0; i < 32; i += 2){
                float w0 = x[i]*inv, w1 = x[i+1]*inv;
                bw += w0*cwv[c0 + i] + w1*cwv[c0 + i + 1];
                v2h hv; hv[0] = (_Float16)(-w0); hv[1] = (_Float16)(-w1);
                *(v2h*)(rp + i) = hv;
            }
#pragma unroll
            for (int m = 1; m <= 4; m <<= 1) bw += __shfl_xor(bw, m, 32);
            if (j == 0) wscv[p*32 + r] = bw;
        }
        if (p < 7) stageKstore(kst, kbuf + (size_t)((p + 1) & 1)*32*LDH, t);
        __syncthreads();

        // GEMM2+3: rs[32 x 64] = Kchunk @ rcb^T + (-scw) @ M2t^T
        {
            v8f acc = {};
#pragma unroll
            for (int k = 0; k < 8; k++){
                v16h b = fragB(rcbh + (size_t)ctg*16*LDH, LDH, k*32);
                acc = WMMA_F16(a[k], b, acc);
            }
#pragma unroll
            for (int k = 0; k < 8; k++){
                v16h a2 = fragA(sch + (size_t)rt*16*LDH, LDH, k*32);
                v16h b2 = fragB(m2t + (size_t)ctg*16*LDH, LDH, k*32);
                acc = WMMA_F16(a2, b2, acc);
            }
            storeAccH(acc, rsh, RST, rt*16, ctg*16);
        }
        __syncthreads();

        // residual row softmax fused into w[s]
        {
            int r = t >> 3, j = t & 7;
            _Float16* rp = rsh + (size_t)r*RST + j*8;
            float y[8];
#pragma unroll
            for (int i = 0; i < 8; i++) y[i] = (float)rp[i];
            float mx = y[0];
#pragma unroll
            for (int i = 1; i < 8; i++) mx = fmaxf(mx, y[i]);
#pragma unroll
            for (int m = 1; m <= 4; m <<= 1) mx = fmaxf(mx, __shfl_xor(mx, m, 32));
            float s = 0.f;
#pragma unroll
            for (int i = 0; i < 8; i++){ y[i] = __expf((y[i] - mx)*0.0625f); s += y[i]; }
#pragma unroll
            for (int m = 1; m <= 4; m <<= 1) s += __shfl_xor(s, m, 32);
            float inv = 1.f / s;
            float rw = 0.f;
#pragma unroll
            for (int i = 0; i < 8; i++) rw += y[i]*inv*rcwv[j*8 + i];
#pragma unroll
            for (int m = 1; m <= 4; m <<= 1) rw += __shfl_xor(rw, m, 32);
            if (j == 0) wscv[p*32 + r] += gate*rw;
        }
        __syncthreads();
    }

    // outputs: effective weights + summary = w @ V (streamed from HBM)
    outEff[(size_t)n*256 + t] = wscv[t] / (1.f + gate);
    float a0 = 0.f, a1 = 0.f, a2 = 0.f, a3 = 0.f;
    for (int s = 0; s < 256; s += 4){
        __builtin_prefetch(Vn + (size_t)((s + 16) & 255)*256 + t, 0, 0);
        a0 += wscv[s + 0]*Vn[(size_t)(s + 0)*256 + t];
        a1 += wscv[s + 1]*Vn[(size_t)(s + 1)*256 + t];
        a2 += wscv[s + 2]*Vn[(size_t)(s + 2)*256 + t];
        a3 += wscv[s + 3]*Vn[(size_t)(s + 3)*256 + t];
    }
    sumOut[(size_t)n*256 + t] = (_Float16)((a0 + a1) + (a2 + a3));
}

// ---- output GEMMs: logits = summary@Wc^T + bc ; recon = summary@Wr^T + br ----
__global__ __launch_bounds__(32)
void kOut(const _Float16* __restrict__ sumh, const _Float16* __restrict__ wch,
          const _Float16* __restrict__ wrh, const float* __restrict__ bc,
          const float* __restrict__ br, float* __restrict__ out){
    int ntile = blockIdx.x, mtile = blockIdx.y;
    bool isl = ntile < (CLSP/16);
    int col0 = isl ? ntile*16 : (ntile - CLSP/16)*16;
    const _Float16* Bb = (isl ? wch : wrh) + (size_t)col0*256;
    const _Float16* Ab = sumh + (size_t)mtile*16*256;
    v8f acc = {};
#pragma unroll
    for (int k = 0; k < 8; k++){
        v16h a = fragA(Ab, 256, k*32);
        v16h b = fragB(Bb, 256, k*32);
        acc = WMMA_F16(a, b, acc);
    }
    int l = threadIdx.x & 31;
    int ncol = l & 15, mo = (l >> 4) << 3;
    int col = col0 + ncol;
    float bias = isl ? ((col < CLS) ? bc[col] : 0.f) : br[col];
#pragma unroll
    for (int v = 0; v < 8; v++){
        int m = mtile*16 + mo + v;
        float val = acc[v] + bias;
        if (isl){ if (col < CLS) out[(size_t)m*CLS + col] = val; }
        else out[(size_t)Nn*CLS + (size_t)m*256 + col] = val;
    }
}

extern "C" void kernel_launch(void* const* d_in, const int* in_sizes, int n_in,
                              void* d_out, int out_size, void* d_ws, size_t ws_size,
                              hipStream_t stream) {
    const float* q   = (const float*)d_in[0];
    const float* K   = (const float*)d_in[1];
    const float* V   = (const float*)d_in[2];
    const float* cb  = (const float*)d_in[3];
    const float* rcb = (const float*)d_in[4];
    const float* wq  = (const float*)d_in[5];
    const float* rl  = (const float*)d_in[6];
    const float* wc  = (const float*)d_in[7];
    const float* bc  = (const float*)d_in[8];
    const float* wr  = (const float*)d_in[9];
    const float* br  = (const float*)d_in[10];
    _Float16* ws = (_Float16*)d_ws;
    float* out = (float*)d_out;
    float* outEff = out + (size_t)Nn*CLS + (size_t)Nn*256;

    kPrep<<<1840, 256, 0, stream>>>(cb, rcb, wq, wc, wr, ws);
    kM2<<<64, 256, 0, stream>>>(cb, rcb, ws);
    kMain<<<Nn, 256, SMEM_BYTES, stream>>>(q, K, V, rl,
        ws + OFF_CB, ws + OFF_RCB, ws + OFF_WQ, ws + OFF_M2,
        outEff, ws + OFF_SUM);
    kOut<<<dim3(CLSP/16 + 16, Nn/16), 32, 0, stream>>>(ws + OFF_SUM, ws + OFF_WC,
        ws + OFF_WR, bc, br, out);
}
